// AdaptiveBlock_1563368096550
// MI455X (gfx1250) — matrix operand order repaired and checked by reading, not compiled, not run on hardware
//
#include <hip/hip_runtime.h>

// ---------------------------------------------------------------------------
// MI455X (gfx1250, wave32) implementation.
// Classifier GEMM (2560x1024)@(1024x50257)^T via v_wmma_f32_16x16x32_f16,
// f32 accumulation. Wmlp converted once to f16 (103 MB -> L2-resident in the
// 192 MB global L2); score stores (515 MB) use non-temporal hints so they do
// not evict the weight slab. Global (AS1) pointers force global_load_b128 /
// global_store (LOADcnt/STOREcnt only) instead of flat ops (which also tick
// DScnt and occupy the LDS path).
// ---------------------------------------------------------------------------

#if defined(__HIP_DEVICE_COMPILE__)
#define GAS __attribute__((address_space(1)))
#else
#define GAS
#endif

typedef __attribute__((ext_vector_type(16))) _Float16 v16h;
typedef __attribute__((ext_vector_type(8)))  _Float16 half8;
typedef __attribute__((ext_vector_type(8)))  float    v8f;

typedef GAS const _Float16* gch_p;
typedef GAS const half8*    gch8_p;
typedef GAS const v16h*     gcv16_p;

#define HD 1024

__device__ __forceinline__ int iminD(int a, int b) { return a < b ? a : b; }

// ---------------------------------------------------------------------------
// Generic f16 WMMA GEMM: C[M,N] (f32, ld=N) = A16[M,K] * B16[N,K]^T
// optionally += existing C, optionally + bias[N].  K multiple of 32.
// Block: 256 threads (8 waves), tile 128(M) x 128(N); wave tile 64x32.
// ntC=1 -> non-temporal C stores (streaming output, preserve L2 for B).
// ---------------------------------------------------------------------------
__global__ __launch_bounds__(256)
void gemm_f16_wmma(const _Float16* __restrict__ A,
                   const _Float16* __restrict__ Bm,
                   float* __restrict__ C,
                   const float* __restrict__ bias,
                   int M, int N, int K, int addC, int ntC)
{
    const int wave = threadIdx.x >> 5;      // 0..7
    const int lane = threadIdx.x & 31;
    const int l16  = lane & 15;
    const int lhi  = lane >> 4;             // 0 or 1

    const int m0 = blockIdx.x * 128 + (wave >> 2) * 64;   // wave M origin
    const int n0 = blockIdx.y * 128 + (wave & 3) * 32;    // wave N origin

    v8f acc[4][2];
#pragma unroll
    for (int mi = 0; mi < 4; ++mi)
#pragma unroll
        for (int ni = 0; ni < 2; ++ni)
            acc[mi][ni] = {};

    // Per-lane global-AS source pointers in contiguous K order (not ISA table order) for both
    // operands (A 16x32: lanes<16 hold K 0..15, lanes>=16 hold 16..31;
    // B 32x16: each lane holds 16 contiguous K halves of one column/row of W).
    gch_p aptr[4];
#pragma unroll
    for (int mi = 0; mi < 4; ++mi) {
        int arow = iminD(m0 + mi * 16 + l16, M - 1);
        aptr[mi] = (gch_p)(A + (size_t)arow * K + lhi * 16);
    }
    gch_p bptr[2];
#pragma unroll
    for (int ni = 0; ni < 2; ++ni) {
        int brow = iminD(n0 + ni * 16 + l16, N - 1);
        bptr[ni] = (gch_p)(Bm + (size_t)brow * K + lhi * 16);
    }

    for (int k = 0; k < K; k += 32) {
        v16h af[4], bf[2];
#pragma unroll
        for (int mi = 0; mi < 4; ++mi) {
            half8 lo = *(gch8_p)(aptr[mi]);         // K = kb .. kb+7
            half8 hi = *(gch8_p)(aptr[mi] + 8);     // K = kb+8 .. kb+15
            af[mi] = __builtin_shufflevector(lo, hi,
                     0,1,2,3,4,5,6,7,8,9,10,11,12,13,14,15);
            aptr[mi] += 32;
        }
#pragma unroll
        for (int ni = 0; ni < 2; ++ni) {
            bf[ni] = *(gcv16_p)(bptr[ni]);
            bptr[ni] += 32;
        }
        // lead the weight stream ~16 K-steps (1 KB) ahead into the caches
        __builtin_prefetch((const void*)(const _Float16*)(bptr[0] + 512), 0, 1);
        __builtin_prefetch((const void*)(const _Float16*)(bptr[1] + 512), 0, 1);
#pragma unroll
        for (int mi = 0; mi < 4; ++mi)
#pragma unroll
            for (int ni = 0; ni < 2; ++ni)
                acc[mi][ni] = __builtin_amdgcn_wmma_f32_16x16x32_f16(
                    false, af[mi], false, bf[ni],
                    (short)0, acc[mi][ni], false, false);
    }

    // Store: C/D layout VGPR r: row = base + lhi*8 + r, col = base + l16.
    GAS float* Cg = (GAS float*)C;
    GAS const float* biasg = (GAS const float*)bias;
#pragma unroll
    for (int mi = 0; mi < 4; ++mi) {
#pragma unroll
        for (int ni = 0; ni < 2; ++ni) {
            int col = n0 + ni * 16 + l16;
            if (col < N) {
                float bv = bias ? biasg[col] : 0.0f;
                int rowbase = m0 + mi * 16 + lhi * 8;
#pragma unroll
                for (int r = 0; r < 8; ++r) {
                    int rr = rowbase + r;
                    if (rr < M) {
                        size_t idx = (size_t)rr * (size_t)N + (size_t)col;
                        float v = acc[mi][ni][r] + bv;
                        if (addC) v += Cg[idx];
                        if (ntC) __builtin_nontemporal_store(v, Cg + idx);
                        else     Cg[idx] = v;
                    }
                }
            }
        }
    }
}

// ---------------------------------------------------------------------------
// fp32 -> f16 conversion
// ---------------------------------------------------------------------------
__global__ __launch_bounds__(256)
void k_cvt(_Float16* __restrict__ dst, const float* __restrict__ src, int n)
{
    int i = blockIdx.x * blockDim.x + threadIdx.x;
    if (i < n) dst[i] = (_Float16)src[i];
}

// h_prev (shifted hiddens) -> f16
__global__ __launch_bounds__(256)
void k_hprev16(_Float16* __restrict__ dst, const float* __restrict__ hid,
               int total, int T)
{
    int i = blockIdx.x * blockDim.x + threadIdx.x;
    if (i >= total) return;
    int t = (i / HD) % T;
    float v = (t == 0) ? 0.0f : hid[i - HD];
    dst[i] = (_Float16)v;
}

// s = sigmoid(G) * tanh(cell)
__global__ __launch_bounds__(256)
void k_sentinel(float* __restrict__ s, const float* __restrict__ G,
                const float* __restrict__ cell, int n)
{
    int i = blockIdx.x * blockDim.x + threadIdx.x;
    if (i >= n) return;
    float g = 1.0f / (1.0f + __expf(-G[i]));
    s[i] = g * tanhf(cell[i]);
}

// out[r,n] = dot(X[r,:1024], W[n,:1024]),  n < 49
__global__ __launch_bounds__(256)
void k_proj49(float* __restrict__ out, const float* __restrict__ X,
              const float* __restrict__ W, int rows)
{
    int i = blockIdx.x * blockDim.x + threadIdx.x;
    int total = rows * 49;
    if (i >= total) return;
    int r = i / 49, n = i - r * 49;
    const float* x = X + (size_t)r * HD;
    const float* w = W + (size_t)n * HD;
    float acc = 0.0f;
    for (int k = 0; k < HD; ++k) acc += x[k] * w[k];
    out[i] = acc;
}

// z[row,k] = sum_j tanh(pV[b,k,j] + g[row,j]) * Wh[j]
__global__ __launch_bounds__(256)
void k_zrow(float* __restrict__ z, const float* __restrict__ pV,
            const float* __restrict__ g, const float* __restrict__ Wh,
            int rows, int T)
{
    int i = blockIdx.x * blockDim.x + threadIdx.x;
    int total = rows * 49;
    if (i >= total) return;
    int row = i / 49, k = i - row * 49;
    int b = row / T;
    const float* pv = pV + ((size_t)b * 49 + k) * 49;
    const float* gr = g + (size_t)row * 49;
    float acc = 0.0f;
    for (int j = 0; j < 49; ++j) acc += tanhf(pv[j] + gr[j]) * Wh[j];
    z[i] = acc;
}

// zs[row] = sum_j tanh(sW[row,j] + g[row,j]) * Wh[j]
__global__ __launch_bounds__(256)
void k_zs(float* __restrict__ zs, const float* __restrict__ sW,
          const float* __restrict__ g, const float* __restrict__ Wh, int rows)
{
    int row = blockIdx.x * blockDim.x + threadIdx.x;
    if (row >= rows) return;
    const float* sw = sW + (size_t)row * 49;
    const float* gr = g + (size_t)row * 49;
    float acc = 0.0f;
    for (int j = 0; j < 49; ++j) acc += tanhf(sw[j] + gr[j]) * Wh[j];
    zs[row] = acc;
}

// z2[b,t,d] = sum_j tanh(pHd[b,d,j] + g2[b,t,j]) * Wh_t[j]
__global__ __launch_bounds__(256)
void k_z2(float* __restrict__ z2, const float* __restrict__ pHd,
          const float* __restrict__ g2, const float* __restrict__ Wh, int total)
{
    int i = blockIdx.x * blockDim.x + threadIdx.x;
    if (i >= total) return;
    int d = i & 127;
    int row = i >> 7;          // b*32 + t
    int b = row >> 5;
    const float* p  = pHd + ((size_t)b * 128 + d) * 49;
    const float* gr = g2 + (size_t)row * 49;
    float acc = 0.0f;
    for (int j = 0; j < 49; ++j) acc += tanhf(p[j] + gr[j]) * Wh[j];
    z2[i] = acc;
}

// Per (b,t): alpha=softmax(z,49), beta=softmax([z,zs])[-1], c=alpha@V[b];
// A16[row,:] = (f16)(beta*s + (1-beta)*c + hid (+ extra))
__global__ __launch_bounds__(128)
void k_chat(_Float16* __restrict__ A16, const float* __restrict__ z,
            const float* __restrict__ zs, const float* __restrict__ s,
            const float* __restrict__ hid, const float* __restrict__ V,
            const float* __restrict__ extra, int T)
{
    __shared__ float sal[49];
    __shared__ float sbeta;
    int row = blockIdx.x;          // b*T + t
    int b = row / T;
    if (threadIdx.x == 0) {
        const float* zr = z + (size_t)row * 49;
        float m49 = -1e30f;
        for (int k = 0; k < 49; ++k) m49 = fmaxf(m49, zr[k]);
        float s49 = 0.0f;
        for (int k = 0; k < 49; ++k) { float e = __expf(zr[k] - m49); sal[k] = e; s49 += e; }
        float inv = 1.0f / s49;
        for (int k = 0; k < 49; ++k) sal[k] *= inv;
        float zsv = zs[row];
        float m50 = fmaxf(m49, zsv);
        float s50 = 0.0f;
        for (int k = 0; k < 49; ++k) s50 += __expf(zr[k] - m50);
        float ez = __expf(zsv - m50);
        sbeta = ez / (s50 + ez);
    }
    __syncthreads();
    float beta = sbeta;
    for (int h = threadIdx.x; h < HD; h += blockDim.x) {
        float c = 0.0f;
        const float* vb = V + (size_t)b * 49 * HD + h;
        for (int k = 0; k < 49; ++k) c += sal[k] * vb[(size_t)k * HD];
        size_t idx = (size_t)row * HD + h;
        float out = beta * s[idx] + (1.0f - beta) * c + hid[idx];
        if (extra) out += extra[idx];
        A16[idx] = (_Float16)out;
    }
}

// Per (b,t): alpha2 = softmax(z2 over 128 d); c2 = alpha2 @ hiddens_des[b]
__global__ __launch_bounds__(128)
void k_c2(float* __restrict__ c2, const float* __restrict__ z2,
          const float* __restrict__ hd)
{
    __shared__ float sal[128];
    int row = blockIdx.x;          // b*32 + t
    int b = row >> 5;
    const float* zr = z2 + (size_t)row * 128;
    if (threadIdx.x < 128) sal[threadIdx.x] = zr[threadIdx.x];
    __syncthreads();
    if (threadIdx.x == 0) {
        float m = -1e30f;
        for (int d = 0; d < 128; ++d) m = fmaxf(m, sal[d]);
        float ssum = 0.0f;
        for (int d = 0; d < 128; ++d) { float e = __expf(sal[d] - m); sal[d] = e; ssum += e; }
        float inv = 1.0f / ssum;
        for (int d = 0; d < 128; ++d) sal[d] *= inv;
    }
    __syncthreads();
    for (int h = threadIdx.x; h < HD; h += blockDim.x) {
        float acc = 0.0f;
        const float* hb = hd + (size_t)b * 128 * HD + h;
        for (int d = 0; d < 128; ++d) acc += sal[d] * hb[(size_t)d * HD];
        c2[(size_t)row * HD + h] = acc;
    }
}

// ---------------------------------------------------------------------------
extern "C" void kernel_launch(void* const* d_in, const int* in_sizes, int n_in,
                              void* d_out, int out_size, void* d_ws, size_t ws_size,
                              hipStream_t stream)
{
    const int Bn = 16, TD = 128, TT = 32, Kk = 49, VOC = 50257;
    const int rows_des = Bn * TD;     // 2048
    const int rows_tit = Bn * TT;     // 512

    const float* x_des  = (const float*)d_in[0];
    const float* x_tit  = (const float*)d_in[1];
    const float* hid_d  = (const float*)d_in[2];
    const float* hid_t  = (const float*)d_in[3];
    const float* cell_d = (const float*)d_in[4];
    const float* cell_t = (const float*)d_in[5];
    const float* Vm     = (const float*)d_in[6];
    const float* Wx     = (const float*)d_in[7];
    const float* Whs    = (const float*)d_in[8];
    const float* Wv_d   = (const float*)d_in[9];
    const float* Wg_d   = (const float*)d_in[10];
    const float* Ws_d   = (const float*)d_in[11];
    const float* Wh_d   = (const float*)d_in[12];
    const float* Wv_t   = (const float*)d_in[13];
    const float* Wg_t   = (const float*)d_in[14];
    const float* Wh_t   = (const float*)d_in[15];
    const float* Wmlp   = (const float*)d_in[16];
    const float* bmlp   = (const float*)d_in[17];

    float* out_des = (float*)d_out;
    float* out_tit = out_des + (size_t)rows_des * VOC;

    // ---- workspace carve-out ------------------------------------------------
    char* p = (char*)d_ws;
    auto alloc = [&](size_t bytes) -> void* {
        void* r = (void*)p;
        p += (bytes + 255) & ~(size_t)255;
        return r;
    };
    _Float16* W16   = (_Float16*)alloc((size_t)VOC * HD * 2);
    _Float16* Wx16  = (_Float16*)alloc((size_t)HD * HD * 2);
    _Float16* Whs16 = (_Float16*)alloc((size_t)HD * HD * 2);
    _Float16* Xd16  = (_Float16*)alloc((size_t)rows_des * HD * 2);
    _Float16* Hpd16 = (_Float16*)alloc((size_t)rows_des * HD * 2);
    _Float16* Xt16  = (_Float16*)alloc((size_t)rows_tit * HD * 2);
    _Float16* Hpt16 = (_Float16*)alloc((size_t)rows_tit * HD * 2);
    _Float16* Ad16  = (_Float16*)alloc((size_t)rows_des * HD * 2);
    _Float16* At16  = (_Float16*)alloc((size_t)rows_tit * HD * 2);
    float* Gdes  = (float*)alloc((size_t)rows_des * HD * 4);
    float* Gtit  = (float*)alloc((size_t)rows_tit * HD * 4);
    float* sdes  = (float*)alloc((size_t)rows_des * HD * 4);
    float* stit  = (float*)alloc((size_t)rows_tit * HD * 4);
    float* pV    = (float*)alloc((size_t)Bn * Kk * Kk * 4);
    float* gdes  = (float*)alloc((size_t)rows_des * Kk * 4);
    float* sWdes = (float*)alloc((size_t)rows_des * Kk * 4);
    float* gtit  = (float*)alloc((size_t)rows_tit * Kk * 4);
    float* sWtit = (float*)alloc((size_t)rows_tit * Kk * 4);
    float* zdes  = (float*)alloc((size_t)rows_des * Kk * 4);
    float* ztit  = (float*)alloc((size_t)rows_tit * Kk * 4);
    float* zsdes = (float*)alloc((size_t)rows_des * 4);
    float* zstit = (float*)alloc((size_t)rows_tit * 4);
    float* pHd   = (float*)alloc((size_t)rows_des * Kk * 4);
    float* g2    = (float*)alloc((size_t)rows_tit * Kk * 4);
    float* z2    = (float*)alloc((size_t)Bn * TT * TD * 4);
    float* c2    = (float*)alloc((size_t)rows_tit * HD * 4);
    (void)ws_size; (void)in_sizes; (void)n_in; (void)out_size;

    auto cdiv = [](long n, long d) { return (int)((n + d - 1) / d); };

    // ---- stage 1: f16 conversions ------------------------------------------
    { int n = VOC * HD;       k_cvt<<<cdiv(n,256),256,0,stream>>>(W16,  Wmlp, n); }
    { int n = HD * HD;        k_cvt<<<cdiv(n,256),256,0,stream>>>(Wx16, Wx,  n); }
    { int n = HD * HD;        k_cvt<<<cdiv(n,256),256,0,stream>>>(Whs16,Whs, n); }
    { int n = rows_des * HD;  k_cvt<<<cdiv(n,256),256,0,stream>>>(Xd16, x_des, n); }
    { int n = rows_tit * HD;  k_cvt<<<cdiv(n,256),256,0,stream>>>(Xt16, x_tit, n); }
    { int n = rows_des * HD;  k_hprev16<<<cdiv(n,256),256,0,stream>>>(Hpd16, hid_d, n, TD); }
    { int n = rows_tit * HD;  k_hprev16<<<cdiv(n,256),256,0,stream>>>(Hpt16, hid_t, n, TT); }

    // ---- stage 2: sentinel gate logits via WMMA ----------------------------
    gemm_f16_wmma<<<dim3(cdiv(rows_des,128), cdiv(HD,128)), 256, 0, stream>>>(
        Xd16, Wx16, Gdes, nullptr, rows_des, HD, HD, 0, 0);
    gemm_f16_wmma<<<dim3(cdiv(rows_des,128), cdiv(HD,128)), 256, 0, stream>>>(
        Hpd16, Whs16, Gdes, nullptr, rows_des, HD, HD, 1, 0);
    gemm_f16_wmma<<<dim3(cdiv(rows_tit,128), cdiv(HD,128)), 256, 0, stream>>>(
        Xt16, Wx16, Gtit, nullptr, rows_tit, HD, HD, 0, 0);
    gemm_f16_wmma<<<dim3(cdiv(rows_tit,128), cdiv(HD,128)), 256, 0, stream>>>(
        Hpt16, Whs16, Gtit, nullptr, rows_tit, HD, HD, 1, 0);

    { int n = rows_des * HD; k_sentinel<<<cdiv(n,256),256,0,stream>>>(sdes, Gdes, cell_d, n); }
    { int n = rows_tit * HD; k_sentinel<<<cdiv(n,256),256,0,stream>>>(stit, Gtit, cell_t, n); }

    // ---- stage 3: K=49 attention projections (fp32, tiny) ------------------
    k_proj49<<<cdiv((long)Bn*Kk*49,256),256,0,stream>>>(pV,    Vm,    Wv_d, Bn*Kk);
    k_proj49<<<cdiv((long)rows_des*49,256),256,0,stream>>>(gdes,  hid_d, Wg_d, rows_des);
    k_proj49<<<cdiv((long)rows_des*49,256),256,0,stream>>>(sWdes, sdes,  Ws_d, rows_des);
    k_proj49<<<cdiv((long)rows_tit*49,256),256,0,stream>>>(gtit,  hid_t, Wg_d, rows_tit);
    k_proj49<<<cdiv((long)rows_tit*49,256),256,0,stream>>>(sWtit, stit,  Ws_d, rows_tit);
    k_proj49<<<cdiv((long)rows_des*49,256),256,0,stream>>>(pHd,   hid_d, Wv_t, rows_des);
    k_proj49<<<cdiv((long)rows_tit*49,256),256,0,stream>>>(g2,    hid_t, Wg_t, rows_tit);

    // ---- stage 4: attention scores -----------------------------------------
    k_zrow<<<cdiv((long)rows_des*49,256),256,0,stream>>>(zdes, pV, gdes, Wh_d, rows_des, TD);
    k_zrow<<<cdiv((long)rows_tit*49,256),256,0,stream>>>(ztit, pV, gtit, Wh_d, rows_tit, TT);
    k_zs<<<cdiv(rows_des,256),256,0,stream>>>(zsdes, sWdes, gdes, Wh_d, rows_des);
    k_zs<<<cdiv(rows_tit,256),256,0,stream>>>(zstit, sWtit, gtit, Wh_d, rows_tit);
    k_z2<<<cdiv((long)Bn*TT*TD,256),256,0,stream>>>(z2, pHd, g2, Wh_t, Bn*TT*TD);

    // ---- stage 5: context mixes, build f16 classifier inputs ---------------
    k_c2<<<rows_tit,128,0,stream>>>(c2, z2, hid_d);
    k_chat<<<rows_des,128,0,stream>>>(Ad16, zdes, zsdes, sdes, hid_d, Vm, nullptr, TD);
    k_chat<<<rows_tit,128,0,stream>>>(At16, ztit, zstit, stit, hid_t, Vm, c2, TT);

    // ---- stage 6: classifier GEMMs (WMMA, W16 L2-resident, NT stores) ------
    gemm_f16_wmma<<<dim3(cdiv(rows_des,128), cdiv(VOC,128)), 256, 0, stream>>>(
        Ad16, W16, out_des, bmlp, rows_des, VOC, HD, 0, 1);
    gemm_f16_wmma<<<dim3(cdiv(rows_tit,128), cdiv(VOC,128)), 256, 0, stream>>>(
        At16, W16, out_tit, bmlp, rows_tit, VOC, HD, 0, 1);
}
